// LinearGlobalIntraContext_42949673372
// MI455X (gfx1250) — matrix operand order repaired and checked by reading, not compiled, not run on hardware
//
#include <hip/hip_runtime.h>
#include <math.h>
#include <stdint.h>

typedef __attribute__((ext_vector_type(16))) __bf16 v16bf;
typedef __attribute__((ext_vector_type(8)))  float  v8f;
typedef __attribute__((ext_vector_type(2)))  float  v2f;

namespace {

constexpr int Bn = 4, Cdim = 64, Hh = 256, Ww = 256, NHEADS = 4;
constexpr int HW  = Hh * Ww;   // 65536
constexpr int W2  = Ww / 2;    // 128
constexpr int HW2 = Hh * W2;   // 32768
constexpr int PAD = 16;        // LDS row padding in halfs (keeps 32B alignment)

__device__ __forceinline__ float gelu_exact(float x) {
  return 0.5f * x * (1.0f + erff(x * 0.7071067811865476f));
}

__device__ __forceinline__ v8f wmma_bf16(v16bf a, v16bf b, v8f c) {
  return __builtin_amdgcn_wmma_f32_16x16x32_bf16(false, a, false, b, (short)0, c,
                                                 false, false);
}

// Async DMA: 16 bytes global -> LDS, tracked by ASYNCcnt (CDNA5 path).
__device__ __forceinline__ void async_copy_b128(unsigned lds_off, const void* gaddr) {
  asm volatile("global_load_async_to_lds_b128 %0, %1, off"
               :: "v"(lds_off), "v"((unsigned long long)(uintptr_t)gaddr)
               : "memory");
}
__device__ __forceinline__ void wait_async0() {
  asm volatile("s_wait_asynccnt 0" ::: "memory");
}

// A fragment (16x32, bf16) gathered from row-major f32 weights W[M][kstride].
// Lane l<16: M=l, K={0..15}; lane>=16: M=l-16, K={16..31}.
__device__ __forceinline__ v16bf load_a_f32rm(const float* __restrict__ W, int kstride,
                                              int oc0, int ic0, int lane) {
  const int m = lane & 15;
  const int base0 = (lane >= 16) ? 16 : 0;
  const float* p = W + (size_t)(oc0 + m) * kstride + ic0 + base0;
  v16bf a;
#pragma unroll
  for (int i = 0; i < 8; ++i) a[i] = (__bf16)p[i];
#pragma unroll
  for (int i = 0; i < 8; ++i) a[8 + i] = (__bf16)p[8 + i];
  return a;
}

// ---------------------------------------------------------------------------
// 1x1 conv as WMMA GEMM: out[b,oc,n] = sum_ic W[oc,ic] * mask(in[b,ic,n]) + bias
// MASKMODE: 0=none, 1=keep (h+w) odd (anchor), 2=keep (h+w) even (nonanchor)
// grid: (HW/128, Cout/16, B), block 256 (8 waves; wave w -> n subtile w)
// ---------------------------------------------------------------------------
template <typename TIN, int CIN, int MASKMODE, bool DO_GELU>
__global__ __launch_bounds__(256) void pw_conv_kernel(const TIN* __restrict__ in,
                                                      const float* __restrict__ Wt,
                                                      const float* __restrict__ bias,
                                                      __bf16* __restrict__ out) {
  __shared__ __align__(32) __bf16 xs[128][CIN + PAD];
  const int b = blockIdx.z;
  const int n0 = blockIdx.x * 128;
  const int oc0 = blockIdx.y * 16;
  const int Cout = gridDim.y * 16;
  const TIN* inb = in + (size_t)b * CIN * HW;

  for (int idx = threadIdx.x; idx < CIN * 128; idx += 256) {
    const int ic = idx >> 7;
    const int col = idx & 127;
    const int n = n0 + col;
    float val = (float)inb[(size_t)ic * HW + n];
    if (MASKMODE != 0) {
      const int hh = n >> 8, wv = n & 255;
      const int par = (hh + wv) & 1;
      if (MASKMODE == 1 ? (par == 0) : (par == 1)) val = 0.0f;
    }
    xs[col][ic] = (__bf16)val;
  }
  __syncthreads();

  const int lane = threadIdx.x & 31;
  const int wave = threadIdx.x >> 5;
  const int nloc = wave * 16 + (lane & 15);
  const int base0b = (lane >= 16) ? 16 : 0;
  v8f acc = {};
#pragma unroll
  for (int kc = 0; kc < CIN / 32; ++kc) {
    v16bf a = load_a_f32rm(Wt, CIN, oc0, kc * 32, lane);
    v16bf bfr = *(const v16bf*)&xs[nloc][kc * 32 + base0b];
    acc = wmma_bf16(a, bfr, acc);
  }
  const int radd = (lane >= 16) ? 8 : 0;
#pragma unroll
  for (int r = 0; r < 8; ++r) {
    const int oc = oc0 + r + radd;
    float v = acc[r] + bias[oc];
    if (DO_GELU) v = gelu_exact(v);
    out[((size_t)b * Cout + oc) * HW + n0 + nloc] = (__bf16)v;
  }
}

// ---------------------------------------------------------------------------
// Depthwise 3x3 (pad 1) + checkerboard squeeze.  sel=0: nonanchor, sel=1: anchor
// out (B,64,H,W2) f32; in (B,64,H,W) bf16
// ---------------------------------------------------------------------------
__global__ __launch_bounds__(256) void dw3x3_squeeze_kernel(const __bf16* __restrict__ in,
                                                            const float* __restrict__ w,
                                                            const float* __restrict__ bias,
                                                            float* __restrict__ out,
                                                            int sel) {
  const unsigned idx = blockIdx.x * 256u + threadIdx.x;  // B*64*H*W2 = 8388608
  const int wi = idx & (W2 - 1);
  const int h = (idx >> 7) & (Hh - 1);
  const int c = (idx >> 15) & 63;
  const int b = idx >> 21;
  const int col = 2 * wi + ((h & 1) ^ sel);
  const __bf16* ip = in + (size_t)(b * Cdim + c) * HW;
  const float* wp = w + c * 9;
  float s = bias[c];
#pragma unroll
  for (int dy = -1; dy <= 1; ++dy) {
    const int hy = h + dy;
    if ((unsigned)hy >= (unsigned)Hh) continue;
#pragma unroll
    for (int dx = -1; dx <= 1; ++dx) {
      const int wx = col + dx;
      if ((unsigned)wx >= (unsigned)Ww) continue;
      s += wp[(dy + 1) * 3 + (dx + 1)] * (float)ip[hy * Ww + wx];
    }
  }
  out[idx] = s;
}

// ---------------------------------------------------------------------------
// Softmax over spatial n=HW2 per (b,c).  grid = B*C, block 256, in-place.
// ---------------------------------------------------------------------------
__global__ __launch_bounds__(256) void softmax_spatial_kernel(float* __restrict__ p_) {
  __shared__ float sm[8];
  __shared__ float ss[8];
  float* p = p_ + (size_t)blockIdx.x * HW2;
  const int tid = threadIdx.x, lane = tid & 31, wave = tid >> 5;
  float m = -3.0e38f;
  for (int i = tid; i < HW2; i += 256) m = fmaxf(m, p[i]);
#pragma unroll
  for (int off = 16; off; off >>= 1) m = fmaxf(m, __shfl_xor(m, off));
  if (lane == 0) sm[wave] = m;
  __syncthreads();
  m = sm[0];
#pragma unroll
  for (int i = 1; i < 8; ++i) m = fmaxf(m, sm[i]);
  float s = 0.0f;
  for (int i = tid; i < HW2; i += 256) s += expf(p[i] - m);
#pragma unroll
  for (int off = 16; off; off >>= 1) s += __shfl_xor(s, off);
  if (lane == 0) ss[wave] = s;
  __syncthreads();
  s = 0.0f;
#pragma unroll
  for (int i = 0; i < 8; ++i) s += ss[i];
  const float inv = 1.0f / s;
  for (int i = tid; i < HW2; i += 256) p[i] = expf(p[i] - m) * inv;
}

// ---------------------------------------------------------------------------
// Softmax over d=16 channels within each head, per position.  In-place on q.
// ---------------------------------------------------------------------------
__global__ __launch_bounds__(256) void softmax_head_kernel(float* __restrict__ q) {
  const unsigned idx = blockIdx.x * 256u + threadIdx.x;  // B*HEADS*HW2 = 524288
  const int pos = idx & (HW2 - 1);
  const int hd = (idx >> 15) & 3;
  const int b = idx >> 17;
  float* p = q + ((size_t)b * Cdim + hd * 16) * HW2 + pos;
  float vals[16];
  float m = -3.0e38f;
#pragma unroll
  for (int d = 0; d < 16; ++d) { vals[d] = p[(size_t)d * HW2]; m = fmaxf(m, vals[d]); }
  float s = 0.0f;
#pragma unroll
  for (int d = 0; d < 16; ++d) { vals[d] = expf(vals[d] - m); s += vals[d]; }
  const float inv = 1.0f / s;
#pragma unroll
  for (int d = 0; d < 16; ++d) p[(size_t)d * HW2] = vals[d] * inv;
}

// ---------------------------------------------------------------------------
// ctx[b,hd,d,e] = sum_n k[b,hd,d,n]*v[b,hd,e,n] via v_wmma_f32_16x16x4_f32.
// grid (8, HEADS, B), block 256 (8 waves); 64 n-splits, atomic f32 accumulate.
// ---------------------------------------------------------------------------
__global__ __launch_bounds__(256) void ctx_kernel(const float* __restrict__ k,
                                                  const float* __restrict__ v,
                                                  float* __restrict__ ctx) {
  const int lane = threadIdx.x & 31, wave = threadIdx.x >> 5;
  const int split = blockIdx.x * 8 + wave;
  const int hd = blockIdx.y, b = blockIdx.z;
  const float* kp = k + ((size_t)b * Cdim + hd * 16) * HW2;
  const float* vp = v + ((size_t)b * Cdim + hd * 16) * HW2;
  const int n0 = split * (HW2 / 64);
  const int row = lane & 15;            // A: M=d row; B: N=e row of v
  const int koff = (lane >= 16) ? 2 : 0;
  v8f acc = {};
  for (int n = n0; n < n0 + HW2 / 64; n += 4) {
    v2f a, bb;
    a[0] = kp[(size_t)row * HW2 + n + koff];
    a[1] = kp[(size_t)row * HW2 + n + koff + 1];
    bb[0] = vp[(size_t)row * HW2 + n + koff];
    bb[1] = vp[(size_t)row * HW2 + n + koff + 1];
    acc = __builtin_amdgcn_wmma_f32_16x16x4_f32(false, a, false, bb, (short)0, acc,
                                                false, false);
  }
  float* cp = ctx + ((size_t)b * NHEADS + hd) * 256;
  const int e = lane & 15;
  const int radd = (lane >= 16) ? 8 : 0;
#pragma unroll
  for (int r = 0; r < 8; ++r) atomicAdd(&cp[(r + radd) * 16 + e], acc[r]);
}

// ---------------------------------------------------------------------------
// att[b, hd*16+e, h, w] = (h+w even) ? sum_d ctx[d,e]*q[d, h, wi] : 0   (bf16)
// grid (H, 64, B), block 256 (w)
// ---------------------------------------------------------------------------
__global__ __launch_bounds__(256) void att_kernel(const float* __restrict__ q,
                                                  const float* __restrict__ ctx,
                                                  __bf16* __restrict__ att) {
  const int w = threadIdx.x, h = blockIdx.x, c = blockIdx.y, b = blockIdx.z;
  const int hd = c >> 4, e = c & 15;
  const size_t oidx = (((size_t)b * Cdim + c) * Hh + h) * Ww + w;
  if (((h + w) & 1) != 0) { att[oidx] = (__bf16)0.0f; return; }
  const int wi = (w - (h & 1)) >> 1;
  const float* qp = q + ((size_t)b * Cdim + hd * 16) * HW2 + (size_t)h * W2 + wi;
  const float* cp = ctx + ((size_t)b * NHEADS + hd) * 256 + e;
  float s = 0.0f;
#pragma unroll
  for (int d = 0; d < 16; ++d) s += cp[d * 16] * qp[(size_t)d * HW2];
  att[oidx] = (__bf16)s;
}

// ---------------------------------------------------------------------------
// Pre-swizzle r_w (128,64,5,5) f32 into WMMA-A-native bf16 layout:
// rwsw[octile(8)][k5(25)][cc(2)][lane(32)][16]
// ---------------------------------------------------------------------------
__global__ __launch_bounds__(256) void swz_rw_kernel(const float* __restrict__ rw,
                                                     __bf16* __restrict__ out) {
  const int idx = blockIdx.x * 256 + threadIdx.x;  // 8*25*2*32*16 = 204800
  if (idx >= 8 * 25 * 2 * 32 * 16) return;
  const int i = idx & 15;
  const int lane = (idx >> 4) & 31;
  const int cc = (idx >> 9) & 1;
  const int t = idx >> 10;
  const int k5 = t % 25;
  const int ot = t / 25;
  const int base0 = (lane >= 16) ? 16 : 0;
  const int kk = base0 + i;
  const int ic = cc * 32 + kk;
  const int oc = ot * 16 + (lane & 15);
  out[idx] = (__bf16)rw[((size_t)oc * 64 + ic) * 25 + k5];
}

// ---------------------------------------------------------------------------
// 5x5 conv (64 -> 128ch, pad 2) implicit GEMM.
// grid (W/64, H, B), block 256: 8 waves = 8 oc-tiles; each wave 4 n-subtiles.
// ---------------------------------------------------------------------------
__global__ __launch_bounds__(256) void conv5x5_kernel(const __bf16* __restrict__ att,
                                                      const __bf16* __restrict__ rwsw,
                                                      const float* __restrict__ rb,
                                                      __bf16* __restrict__ attn) {
  __shared__ __align__(32) __bf16 xs[5][68][64 + PAD];
  const int w0 = blockIdx.x * 64, h = blockIdx.y, b = blockIdx.z;
  const __bf16* ab = att + (size_t)b * Cdim * HW;
  for (int idx = threadIdx.x; idx < 5 * 64 * 68; idx += 256) {
    const int col = idx % 68;
    const int t = idx / 68;
    const int ic = t & 63;
    const int r = t >> 6;
    const int hy = h + r - 2;
    const int wx = w0 + col - 2;
    float val = 0.0f;
    if ((unsigned)hy < (unsigned)Hh && (unsigned)wx < (unsigned)Ww)
      val = (float)ab[((size_t)ic * Hh + hy) * Ww + wx];
    xs[r][col][ic] = (__bf16)val;
  }
  __syncthreads();

  const int lane = threadIdx.x & 31, wave = threadIdx.x >> 5;
  const int nlo = lane & 15;
  const int base0b = (lane >= 16) ? 16 : 0;
  const __bf16* wsw = rwsw + (size_t)wave * 25 * 2 * 32 * 16;
  v8f acc[4] = {};
#pragma unroll
  for (int kh = 0; kh < 5; ++kh)
    for (int kw = 0; kw < 5; ++kw)
#pragma unroll
      for (int cc = 0; cc < 2; ++cc) {
        v16bf a = *(const v16bf*)(wsw + (size_t)(((kh * 5 + kw) * 2 + cc) * 32 + lane) * 16);
#pragma unroll
        for (int ns = 0; ns < 4; ++ns) {
          v16bf bfr = *(const v16bf*)&xs[kh][ns * 16 + nlo + kw][cc * 32 + base0b];
          acc[ns] = wmma_bf16(a, bfr, acc[ns]);
        }
      }
  const int radd = (lane >= 16) ? 8 : 0;
#pragma unroll
  for (int ns = 0; ns < 4; ++ns)
#pragma unroll
    for (int r = 0; r < 8; ++r) {
      const int oc = wave * 16 + r + radd;
      const int w = w0 + ns * 16 + nlo;
      attn[(((size_t)b * 128 + oc) * Hh + h) * Ww + w] = (__bf16)(acc[ns][r] + rb[oc]);
    }
}

// ---------------------------------------------------------------------------
// Fused tail: m1 halo async-DMA'd to LDS (64-channel chunks), m2 =
// gelu(dw3x3+b2) staged in LDS, then m3 (256->128) WMMA GEMM + bias +
// residual attention -> f32 output.  grid (W/64, H, B), block 256.
// OOB halo segments stay inside the workspace allocation and are never read
// (tap loop bounds-checks with global coordinates).
// ---------------------------------------------------------------------------
__global__ __launch_bounds__(256) void mlp_tail_kernel(const __bf16* __restrict__ m1,
                                                       const float* __restrict__ m2w,
                                                       const float* __restrict__ m2b,
                                                       const float* __restrict__ m3w,
                                                       const float* __restrict__ m3b,
                                                       const __bf16* __restrict__ attn,
                                                       float* __restrict__ out) {
  __shared__ __align__(32) __bf16 xs[64][256 + PAD];   // m2 tile [n=64][ic=256]
  __shared__ __align__(16) __bf16 ms[3][64][72];       // m1 halo stage per 64ch chunk
  const int w0 = blockIdx.x * 64, h = blockIdx.y, b = blockIdx.z;
  const __bf16* m1p = m1 + (size_t)b * 256 * HW;
  const int col = threadIdx.x & 63;
  const int w = w0 + col;

  for (int icc = 0; icc < 4; ++icc) {
    // Async-stage 3 rows x 64 channels x 72 cols (b128 granules, ASYNCcnt).
    for (int idx = threadIdx.x; idx < 3 * 64 * 9; idx += 256) {
      const int seg = idx % 9;
      const int t = idx / 9;
      const int icl = t & 63;
      const int r = t >> 6;
      const int hy = h + r - 1;
      const __bf16* g =
          m1p + ((size_t)(icc * 64 + icl) * Hh + hy) * Ww + (w0 - 4 + seg * 8);
      async_copy_b128((unsigned)(uintptr_t)&ms[r][icl][seg * 8], g);
    }
    wait_async0();
    __syncthreads();
    // depthwise 3x3 + gelu for this channel chunk, out of LDS
    for (int icl = (threadIdx.x >> 6); icl < 64; icl += 4) {
      const int ic = icc * 64 + icl;
      const float* wp = m2w + ic * 9;
      float s = m2b[ic];
#pragma unroll
      for (int dy = -1; dy <= 1; ++dy) {
        const int hy = h + dy;
        if ((unsigned)hy >= (unsigned)Hh) continue;
#pragma unroll
        for (int dx = -1; dx <= 1; ++dx) {
          const int wx = w + dx;
          if ((unsigned)wx >= (unsigned)Ww) continue;
          s += wp[(dy + 1) * 3 + (dx + 1)] * (float)ms[dy + 1][icl][col + 4 + dx];
        }
      }
      xs[col][ic] = (__bf16)gelu_exact(s);
    }
    __syncthreads();
  }

  const int lane = threadIdx.x & 31, wave = threadIdx.x >> 5;
  const int oc0 = wave * 16;
  const int nlo = lane & 15;
  const int base0b = (lane >= 16) ? 16 : 0;
  v8f acc[4] = {};
#pragma unroll
  for (int kc = 0; kc < 8; ++kc) {
    v16bf a = load_a_f32rm(m3w, 256, oc0, kc * 32, lane);
#pragma unroll
    for (int ns = 0; ns < 4; ++ns) {
      v16bf bfr = *(const v16bf*)&xs[ns * 16 + nlo][kc * 32 + base0b];
      acc[ns] = wmma_bf16(a, bfr, acc[ns]);
    }
  }
  const int radd = (lane >= 16) ? 8 : 0;
#pragma unroll
  for (int ns = 0; ns < 4; ++ns)
#pragma unroll
    for (int r = 0; r < 8; ++r) {
      const int oc = oc0 + r + radd;
      const size_t oi = (((size_t)b * 128 + oc) * Hh + h) * Ww + (w0 + ns * 16 + nlo);
      out[oi] = acc[ns][r] + m3b[oc] + (float)attn[oi];
    }
}

}  // namespace

extern "C" void kernel_launch(void* const* d_in, const int* in_sizes, int n_in,
                              void* d_out, int out_size, void* d_ws, size_t ws_size,
                              hipStream_t stream) {
  (void)in_sizes; (void)n_in; (void)out_size; (void)ws_size;
  const float* x1  = (const float*)d_in[0];
  const float* x2  = (const float*)d_in[1];
  const float* q1w = (const float*)d_in[2];
  const float* q1b = (const float*)d_in[3];
  const float* q2w = (const float*)d_in[4];
  const float* q2b = (const float*)d_in[5];
  const float* k1w = (const float*)d_in[6];
  const float* k1b = (const float*)d_in[7];
  const float* k2w = (const float*)d_in[8];
  const float* k2b = (const float*)d_in[9];
  const float* v1w = (const float*)d_in[10];
  const float* v1b = (const float*)d_in[11];
  const float* v2w = (const float*)d_in[12];
  const float* v2b = (const float*)d_in[13];
  const float* rw  = (const float*)d_in[14];
  const float* rb  = (const float*)d_in[15];
  const float* m1w = (const float*)d_in[16];
  const float* m1b_ = (const float*)d_in[17];
  const float* m2w = (const float*)d_in[18];
  const float* m2b = (const float*)d_in[19];
  const float* m3w = (const float*)d_in[20];
  const float* m3b = (const float*)d_in[21];
  float* out = (float*)d_out;

  const size_t MB = 1ull << 20;
  char* ws = (char*)d_ws;
  // Lifetime-overlapped workspace layout (~225 MB total):
  __bf16* qpre = (__bf16*)(ws + 0 * MB);     // [0,32)
  __bf16* kpre = (__bf16*)(ws + 32 * MB);    // [32,64)
  __bf16* vpre = (__bf16*)(ws + 64 * MB);    // [64,96)
  float*  qsq  = (float*)(ws + 96 * MB);     // [96,128)
  float*  ksq  = (float*)(ws + 128 * MB);    // [128,160)
  float*  vsq  = (float*)(ws + 160 * MB);    // [160,192)
  float*  ctx  = (float*)(ws + 192 * MB);    // 16KB
  __bf16* att  = (__bf16*)(ws + 0 * MB);     // reuse qpre
  __bf16* attn = (__bf16*)(ws + 32 * MB);    // reuse kpre+vpre (64MB)
  __bf16* m1d  = (__bf16*)(ws + 96 * MB);    // reuse q/k/v+ctx (128MB)
  __bf16* rwsw = (__bf16*)(ws + 224 * MB);   // 0.4MB

  const dim3 blk(256);

  // QKV 1x1 convs (masked implicit GEMM, WMMA bf16)
  pw_conv_kernel<float, 64, 2, false><<<dim3(512, 4, 4), blk, 0, stream>>>(x1, q1w, q1b, qpre);
  pw_conv_kernel<float, 64, 1, false><<<dim3(512, 4, 4), blk, 0, stream>>>(x1, k1w, k1b, kpre);
  pw_conv_kernel<float, 64, 0, false><<<dim3(512, 4, 4), blk, 0, stream>>>(x2, v1w, v1b, vpre);

  // depthwise 3x3 + checkerboard squeeze
  const int nsq = Bn * Cdim * Hh * W2;  // 8388608
  dw3x3_squeeze_kernel<<<dim3(nsq / 256), blk, 0, stream>>>(qpre, q2w, q2b, qsq, 0);
  dw3x3_squeeze_kernel<<<dim3(nsq / 256), blk, 0, stream>>>(kpre, k2w, k2b, ksq, 1);
  dw3x3_squeeze_kernel<<<dim3(nsq / 256), blk, 0, stream>>>(vpre, v2w, v2b, vsq, 1);

  // softmaxes
  softmax_spatial_kernel<<<dim3(Bn * Cdim), blk, 0, stream>>>(ksq);
  softmax_head_kernel<<<dim3(Bn * NHEADS * HW2 / 256), blk, 0, stream>>>(qsq);

  // ctx = k v^T per head (f32 WMMA + atomic accumulation)
  hipMemsetAsync(ctx, 0, (size_t)Bn * NHEADS * 256 * sizeof(float), stream);
  ctx_kernel<<<dim3(8, NHEADS, Bn), blk, 0, stream>>>(ksq, vsq, ctx);

  // att = ctx^T q (nonanchor-unsqueezed, bf16)
  att_kernel<<<dim3(Hh, Cdim, Bn), blk, 0, stream>>>(qsq, ctx, att);

  // 5x5 conv -> attention (bf16)
  swz_rw_kernel<<<dim3(800), blk, 0, stream>>>(rw, rwsw);
  conv5x5_kernel<<<dim3(Ww / 64, Hh, Bn), blk, 0, stream>>>(att, rwsw, rb, attn);

  // MLP: m1 (1x1 128->256 + GELU), then fused async-staged dw3x3+GELU +
  // 1x1 256->128 + residual
  pw_conv_kernel<__bf16, 128, 0, true><<<dim3(512, 16, 4), blk, 0, stream>>>(attn, m1w, m1b_, m1d);
  mlp_tail_kernel<<<dim3(Ww / 64, Hh, Bn), blk, 0, stream>>>(m1d, m2w, m2b, m3w, m3b, attn, out);
}
